// LinearSelfAttention3D_31061203484910
// MI455X (gfx1250) — matrix operand order repaired and checked by reading, not compiled, not run on hardware
//
#include <hip/hip_runtime.h>
#include <hip/hip_bf16.h>
#include <math.h>

typedef __attribute__((ext_vector_type(16))) __bf16 v16bf;
typedef __attribute__((ext_vector_type(8)))  __bf16 v8bf;
typedef __attribute__((ext_vector_type(8)))  float  v8f;
typedef __attribute__((ext_vector_type(4)))  float  v4f;

#define B_   16
#define C_   256
#define S_   4096
#define MD   512
#define NH   8
#define HD   64
#define NMEM 4
#define QKV_ROWS 1536
#define SCALE_ 0.125f   // 64^-0.5
#define EPS_  1e-5f

// workspace offsets (bytes)
#define OFF_QKV   0ull
#define OFF_QSM   201326592ull   // qkv bf16: 16*1536*4096*2
#define OFF_RMAX  268435456ull   // + qsm bf16: 16*512*4096*2
#define OFF_RSUM  268468224ull   // + 8192*4
#define OFF_CTX   268500992ull   // + 8192*4
#define OFF_OUT   270598144ull   // + ctx f32: 16*8*64*64*4

// ---------------- WMMA fragment helpers (wave32; this file's contiguous K order) ----------------
// A 16x32 bf16: lanes 0-15 M=lane; laneHalf=0 -> K 0..15, laneHalf=1 -> K 16..31.
__device__ __forceinline__ v16bf load_a_frag(const __bf16* ldsA, int row0,
                                             int stride, int k0, int lane) {
    int m = lane & 15, kh = lane >> 4;
    const __bf16* p = ldsA + (size_t)(row0 + m) * stride + k0;
    v8bf lo = *(const v8bf*)(p + kh * 16);
    v8bf hi = *(const v8bf*)(p + kh * 16 + 8);
    v16bf a;
#pragma unroll
    for (int i = 0; i < 8; ++i) { a[i] = lo[i]; a[8 + i] = hi[i]; }
    return a;
}
// B 32x16 bf16: lane&15 = N; laneHalf=0 -> K 0..15, laneHalf=1 -> K 16..31.
// ldsBt is stored [N][K] so the fragment is one contiguous 32-byte read.
__device__ __forceinline__ v16bf load_b_frag(const __bf16* ldsBt, int ncol0,
                                             int stride, int lane) {
    int n = lane & 15, kh = lane >> 4;
    return *(const v16bf*)(ldsBt + (size_t)(ncol0 + n) * stride + kh * 16);
}

// Async global->LDS copy of 16 bytes per lane (ASYNCcnt-tracked, no VGPR data).
__device__ __forceinline__ void async_copy_b128(const __bf16* lds_dst,
                                                const __bf16* gsrc) {
    unsigned lds_off = (unsigned)(uintptr_t)lds_dst;  // flat LDS aperture: low 32 = LDS offset
    asm volatile("global_load_async_to_lds_b128 %0, %1, off"
                 :: "v"(lds_off), "v"(gsrc) : "memory");
}
__device__ __forceinline__ void wait_asynccnt0() {
    asm volatile("s_wait_asynccnt 0x0" ::: "memory");
}

// ---------------- Stage 1: qkv = w_qkv @ x[b] + bias -> bf16 ----------------
// Block tile 64(M) x 128(N), 8 waves, each wave a 16x64 slab (4 WMMAs / K-step).
__global__ __launch_bounds__(256) void qkv_gemm(const float* __restrict__ x,
                                                const float* __restrict__ wqkv,
                                                const float* __restrict__ bqkv,
                                                __bf16* __restrict__ qkv) {
    __shared__ __attribute__((aligned(32))) __bf16 ldsA[64 * 32];
    __shared__ __attribute__((aligned(32))) __bf16 ldsBt[128 * 32];
    int tid = threadIdx.x, lane = tid & 31, wave = tid >> 5;
    int wM = (wave & 3) * 16, wN = (wave >> 2) * 64;
    int n0 = blockIdx.x * 128, m0 = blockIdx.y * 64, b = blockIdx.z;
    const float* xb = x + (size_t)b * C_ * S_;
    v8f acc[4] = {};
    for (int k0 = 0; k0 < C_; k0 += 32) {
        {   // A: w_qkv [64 m][32 k], f32 -> bf16
            int lin = tid * 8, ar = lin >> 5, ac = lin & 31;
            const float* gp = wqkv + (size_t)(m0 + ar) * C_ + k0 + ac;
            v4f f0 = *(const v4f*)gp, f1 = *(const v4f*)(gp + 4);
            __bf16* dp = ldsA + ar * 32 + ac;
#pragma unroll
            for (int i = 0; i < 4; ++i) { dp[i] = (__bf16)f0[i]; dp[4 + i] = (__bf16)f1[i]; }
        }
        {   // B: x [32 k][128 n] -> ldsBt[n][k], f32 -> bf16 (transposed staging)
            int lin = tid * 16, kr = lin >> 7, nc = lin & 127;
            const float* gp = xb + (size_t)(k0 + kr) * S_ + n0 + nc;
#pragma unroll
            for (int q = 0; q < 4; ++q) {
                v4f f = *(const v4f*)(gp + q * 4);
#pragma unroll
                for (int i = 0; i < 4; ++i)
                    ldsBt[(nc + q * 4 + i) * 32 + kr] = (__bf16)f[i];
            }
        }
        __syncthreads();
        v16bf a = load_a_frag(ldsA, wM, 32, 0, lane);
        v16bf bt[4];
#pragma unroll
        for (int t = 0; t < 4; ++t) bt[t] = load_b_frag(ldsBt, wN + t * 16, 32, lane);
#pragma unroll
        for (int t = 0; t < 4; ++t)
            acc[t] = __builtin_amdgcn_wmma_f32_16x16x32_bf16(false, a, false, bt[t],
                                                             (short)0, acc[t], false, false);
        __syncthreads();
    }
    int kh = lane >> 4, nn = lane & 15;
#pragma unroll
    for (int r = 0; r < 8; ++r) {
        int o = m0 + wM + kh * 8 + r;
        float bias = bqkv[o];
        size_t base = ((size_t)b * QKV_ROWS + o) * S_ + n0 + wN + nn;
#pragma unroll
        for (int t = 0; t < 4; ++t)
            qkv[base + t * 16] = (__bf16)(acc[t][r] + bias);
    }
}

// ---------------- Stage 2: q softmax over head_dim, * SCALE ----------------
__global__ __launch_bounds__(256) void q_softmax(const __bf16* __restrict__ qkv,
                                                 __bf16* __restrict__ qsm) {
    int g = blockIdx.x * 256 + threadIdx.x;      // g = ((b*8+head)<<12) | s
    int s = g & (S_ - 1);
    int head = (g >> 12) & (NH - 1);
    int b = g >> 15;
    const __bf16* qp = qkv + ((size_t)b * QKV_ROWS + head * HD) * S_ + s;
    float vals[HD];
    float mx = -1e30f;
#pragma unroll
    for (int d = 0; d < HD; ++d) { vals[d] = (float)qp[(size_t)d * S_]; mx = fmaxf(mx, vals[d]); }
    float sum = 0.f;
#pragma unroll
    for (int d = 0; d < HD; ++d) { vals[d] = __expf(vals[d] - mx); sum += vals[d]; }
    float inv = SCALE_ / sum;
    __bf16* op = qsm + ((size_t)(b * NH + head) * HD) * S_ + s;
#pragma unroll
    for (int d = 0; d < HD; ++d) op[(size_t)d * S_] = (__bf16)(vals[d] * inv);
}

// ---------------- Stage 3: per-row max / sum-exp for k (incl. memory) --------
__global__ __launch_bounds__(256) void k_rowstats(const __bf16* __restrict__ qkv,
                                                  const float* __restrict__ memkv,
                                                  float* __restrict__ rmax,
                                                  float* __restrict__ rsum) {
    __shared__ float red[256];
    int rid = blockIdx.x;                 // rid = b*512 + head*64 + d
    int tid = threadIdx.x;
    int b = rid >> 9, hd = rid & 511;
    const __bf16* kp = qkv + ((size_t)b * QKV_ROWS + MD + hd) * S_;
    float mx = -1e30f;
    for (int s = tid; s < S_; s += 256) mx = fmaxf(mx, (float)kp[s]);
    red[tid] = mx;
    __syncthreads();
    for (int off = 128; off > 0; off >>= 1) {
        if (tid < off) red[tid] = fmaxf(red[tid], red[tid + off]);
        __syncthreads();
    }
    if (tid == 0) {
        float m2 = red[0];
        for (int m = 0; m < NMEM; ++m) m2 = fmaxf(m2, memkv[hd * NMEM + m]);
        red[0] = m2;
    }
    __syncthreads();
    mx = red[0];
    __syncthreads();
    float sm = 0.f;
    for (int s = tid; s < S_; s += 256) sm += __expf((float)kp[s] - mx);
    red[tid] = sm;
    __syncthreads();
    for (int off = 128; off > 0; off >>= 1) {
        if (tid < off) red[tid] += red[tid + off];
        __syncthreads();
    }
    if (tid == 0) {
        float tot = red[0];
        for (int m = 0; m < NMEM; ++m) tot += __expf(memkv[hd * NMEM + m] - mx);
        rmax[rid] = mx;
        rsum[rid] = tot;
    }
}

// ---------------- Stage 4a: zero the context accumulator ----------------
__global__ __launch_bounds__(256) void ctx_zero(float* __restrict__ ctx) {
    int i = (blockIdx.x * 256 + threadIdx.x) * 4;
    v4f z = {0.f, 0.f, 0.f, 0.f};
    *(v4f*)(ctx + i) = z;
}

// ---------------- Stage 4b: context += softmax(k) @ v^T  (K split 4-way) -----
// v tile staging is a pure bf16 copy -> GLOBAL_LOAD_ASYNC_TO_LDS_B128.
__global__ __launch_bounds__(256) void context_gemm(const __bf16* __restrict__ qkv,
                                                    const float* __restrict__ memkv,
                                                    const float* __restrict__ rmax,
                                                    const float* __restrict__ rsum,
                                                    float* __restrict__ ctx) {
    __shared__ __attribute__((aligned(32))) __bf16 ldsA[64 * 32];
    __shared__ __attribute__((aligned(32))) __bf16 ldsBt[64 * 32];
    int tid = threadIdx.x, lane = tid & 31, wave = tid >> 5;
    int wM = (wave & 3) * 16, wN = (wave >> 2) * 32;
    int bh = blockIdx.x;                  // b*8 + head
    int kchunk = blockIdx.y;              // K split: [kchunk*1024, +1024)
    int b = bh >> 3, head = bh & 7;
    const __bf16* kbase = qkv + ((size_t)b * QKV_ROWS + MD + head * HD) * S_;
    const __bf16* vbase = qkv + ((size_t)b * QKV_ROWS + 2 * MD + head * HD) * S_;
    int lin = tid * 8, ir = lin >> 5, kc = lin & 31;
    float rm = rmax[bh * HD + ir];
    float rs = 1.f / rsum[bh * HD + ir];
    v8f c0 = {}, c1 = {};
    int kbeg = kchunk * (S_ / 4), kend = kbeg + (S_ / 4);
    for (int k0 = kbeg; k0 < kend; k0 += 32) {
        // Bt: v rows are already [j][s] = [N][K]: async DMA straight into LDS
        async_copy_b128(ldsBt + ir * 32 + kc, vbase + (size_t)ir * S_ + k0 + kc);
        {   // A: softmax-normalize k row on the fly -> bf16
            v8bf kv = *(const v8bf*)(kbase + (size_t)ir * S_ + k0 + kc);
            __bf16* dp = ldsA + ir * 32 + kc;
#pragma unroll
            for (int i = 0; i < 8; ++i) dp[i] = (__bf16)(__expf((float)kv[i] - rm) * rs);
        }
        wait_asynccnt0();
        __syncthreads();
        v16bf a  = load_a_frag(ldsA, wM, 32, 0, lane);
        v16bf b0 = load_b_frag(ldsBt, wN, 32, lane);
        v16bf b1 = load_b_frag(ldsBt, wN + 16, 32, lane);
        c0 = __builtin_amdgcn_wmma_f32_16x16x32_bf16(false, a, false, b0, (short)0, c0, false, false);
        c1 = __builtin_amdgcn_wmma_f32_16x16x32_bf16(false, a, false, b1, (short)0, c1, false, false);
        __syncthreads();
    }
    int kh = lane >> 4, nn = lane & 15;
    const float* mk = memkv + head * HD * NMEM;                    // mem_kv[0][head]
    const float* mv = memkv + (NH * HD * NMEM) + head * HD * NMEM; // mem_kv[1][head]
#pragma unroll
    for (int r = 0; r < 8; ++r) {
        int i = wM + kh * 8 + r;
        int j0 = wN + nn, j1 = wN + 16 + nn;
        float a0 = c0[r], a1 = c1[r];
        if (kchunk == 0) {   // memory-slot columns added exactly once
            float rm_i = rmax[bh * HD + i], rs_i = 1.f / rsum[bh * HD + i];
            for (int m = 0; m < NMEM; ++m) {
                float ke = __expf(mk[i * NMEM + m] - rm_i) * rs_i;
                a0 += ke * mv[j0 * NMEM + m];
                a1 += ke * mv[j1 * NMEM + m];
            }
        }
        atomicAdd(&ctx[((size_t)bh * HD + i) * HD + j0], a0);
        atomicAdd(&ctx[((size_t)bh * HD + i) * HD + j1], a1);
    }
}

// ---------------- Stage 5: out = context^T @ q_sm -> bf16 ----------------
// Block tile 64(M=j) x 128(N=s); full A (64x64) staged once.
__global__ __launch_bounds__(256) void attn_out(const float* __restrict__ ctx,
                                                const __bf16* __restrict__ qsm,
                                                __bf16* __restrict__ outb) {
    __shared__ __attribute__((aligned(32))) __bf16 ldsA[64 * 64];  // A[j][i]=ctx[i][j]
    __shared__ __attribute__((aligned(32))) __bf16 ldsBt[128 * 32];
    int tid = threadIdx.x, lane = tid & 31, wave = tid >> 5;
    int wM = (wave & 3) * 16, wN = (wave >> 2) * 64;
    int n0 = blockIdx.x * 128;
    int bh = blockIdx.y;
    const float* cbase = ctx + (size_t)bh * HD * HD;
    const __bf16* qbase = qsm + (size_t)bh * HD * S_;
#pragma unroll
    for (int e = 0; e < 16; ++e) {        // stage full A once (transpose+convert)
        int idx = tid * 16 + e;
        int i = idx >> 6, j = idx & 63;
        ldsA[j * 64 + i] = (__bf16)cbase[idx];
    }
    v8f acc[4] = {};
    for (int kk = 0; kk < 2; ++kk) {
        int k0 = kk * 32;
        {   // Bt: q_sm [32 k][128 n] -> ldsBt[n][k]
            int lin = tid * 16, kr = lin >> 7, nc = lin & 127;
            const __bf16* gp = qbase + (size_t)(k0 + kr) * S_ + n0 + nc;
            v8bf q0 = *(const v8bf*)gp;
            v8bf q1 = *(const v8bf*)(gp + 8);
#pragma unroll
            for (int i2 = 0; i2 < 8; ++i2) {
                ldsBt[(nc + i2) * 32 + kr]     = q0[i2];
                ldsBt[(nc + 8 + i2) * 32 + kr] = q1[i2];
            }
        }
        __syncthreads();
        v16bf a = load_a_frag(ldsA, wM, 64, k0, lane);
        v16bf bt[4];
#pragma unroll
        for (int t = 0; t < 4; ++t) bt[t] = load_b_frag(ldsBt, wN + t * 16, 32, lane);
#pragma unroll
        for (int t = 0; t < 4; ++t)
            acc[t] = __builtin_amdgcn_wmma_f32_16x16x32_bf16(false, a, false, bt[t],
                                                             (short)0, acc[t], false, false);
        __syncthreads();
    }
    int kh = lane >> 4, nn = lane & 15;
#pragma unroll
    for (int r = 0; r < 8; ++r) {
        int j = wM + kh * 8 + r;
        size_t row = (size_t)bh * HD + j;   // == b*512 + head*64 + j
#pragma unroll
        for (int t = 0; t < 4; ++t)
            outb[row * S_ + n0 + wN + t * 16 + nn] = (__bf16)acc[t][r];
    }
}

// ---------------- Stage 6: y = BN(w_out @ out + b_out) -> f32 ----------------
__global__ __launch_bounds__(256) void out_proj_bn(const __bf16* __restrict__ outb,
                                                   const float* __restrict__ wout,
                                                   const float* __restrict__ bout,
                                                   const float* __restrict__ gamma,
                                                   const float* __restrict__ beta,
                                                   const float* __restrict__ mean,
                                                   const float* __restrict__ var,
                                                   float* __restrict__ y) {
    __shared__ __attribute__((aligned(32))) __bf16 ldsA[64 * 32];
    __shared__ __attribute__((aligned(32))) __bf16 ldsBt[128 * 32];
    int tid = threadIdx.x, lane = tid & 31, wave = tid >> 5;
    int wM = (wave & 3) * 16, wN = (wave >> 2) * 64;
    int n0 = blockIdx.x * 128, m0 = blockIdx.y * 64, b = blockIdx.z;
    const __bf16* ob = outb + (size_t)b * MD * S_;
    v8f acc[4] = {};
    for (int k0 = 0; k0 < MD; k0 += 32) {
        {   // A: w_out [64 m][32 k], f32 -> bf16
            int lin = tid * 8, ar = lin >> 5, ac = lin & 31;
            const float* gp = wout + (size_t)(m0 + ar) * MD + k0 + ac;
            v4f f0 = *(const v4f*)gp, f1 = *(const v4f*)(gp + 4);
            __bf16* dp = ldsA + ar * 32 + ac;
#pragma unroll
            for (int i = 0; i < 4; ++i) { dp[i] = (__bf16)f0[i]; dp[4 + i] = (__bf16)f1[i]; }
        }
        {   // Bt: out [32 k][128 n] -> ldsBt[n][k] (already bf16, transpose)
            int lin = tid * 16, kr = lin >> 7, nc = lin & 127;
            const __bf16* gp = ob + (size_t)(k0 + kr) * S_ + n0 + nc;
            v8bf o0 = *(const v8bf*)gp;
            v8bf o1 = *(const v8bf*)(gp + 8);
#pragma unroll
            for (int i = 0; i < 8; ++i) {
                ldsBt[(nc + i) * 32 + kr]     = o0[i];
                ldsBt[(nc + 8 + i) * 32 + kr] = o1[i];
            }
        }
        __syncthreads();
        v16bf a = load_a_frag(ldsA, wM, 32, 0, lane);
        v16bf bt[4];
#pragma unroll
        for (int t = 0; t < 4; ++t) bt[t] = load_b_frag(ldsBt, wN + t * 16, 32, lane);
#pragma unroll
        for (int t = 0; t < 4; ++t)
            acc[t] = __builtin_amdgcn_wmma_f32_16x16x32_bf16(false, a, false, bt[t],
                                                             (short)0, acc[t], false, false);
        __syncthreads();
    }
    int kh = lane >> 4, nn = lane & 15;
#pragma unroll
    for (int r = 0; r < 8; ++r) {
        int o = m0 + wM + kh * 8 + r;
        float inv = rsqrtf(var[o] + EPS_) * gamma[o];
        float base = beta[o] - mean[o] * inv;
        size_t row = ((size_t)b * C_ + o) * S_ + n0 + wN + nn;
#pragma unroll
        for (int t = 0; t < 4; ++t)
            y[row + t * 16] = (acc[t][r] + bout[o]) * inv + base;
    }
}

extern "C" void kernel_launch(void* const* d_in, const int* in_sizes, int n_in,
                              void* d_out, int out_size, void* d_ws, size_t ws_size,
                              hipStream_t stream) {
    const float* x     = (const float*)d_in[0];
    const float* wqkv  = (const float*)d_in[1];
    const float* bqkv  = (const float*)d_in[2];
    const float* memkv = (const float*)d_in[3];
    const float* wout  = (const float*)d_in[4];
    const float* bout  = (const float*)d_in[5];
    const float* gamma = (const float*)d_in[6];
    const float* beta  = (const float*)d_in[7];
    const float* mean  = (const float*)d_in[8];
    const float* var   = (const float*)d_in[9];
    float* y = (float*)d_out;

    char* ws = (char*)d_ws;
    __bf16* qkv  = (__bf16*)(ws + OFF_QKV);
    __bf16* qsm  = (__bf16*)(ws + OFF_QSM);
    float*  rmax = (float*)(ws + OFF_RMAX);
    float*  rsum = (float*)(ws + OFF_RSUM);
    float*  ctx  = (float*)(ws + OFF_CTX);
    __bf16* outb = (__bf16*)(ws + OFF_OUT);

    qkv_gemm    <<<dim3(S_ / 128, QKV_ROWS / 64, B_), 256, 0, stream>>>(x, wqkv, bqkv, qkv);
    q_softmax   <<<dim3(B_ * NH * S_ / 256), 256, 0, stream>>>(qkv, qsm);
    k_rowstats  <<<dim3(B_ * NH * HD), 256, 0, stream>>>(qkv, memkv, rmax, rsum);
    ctx_zero    <<<dim3(B_ * NH * HD * HD / 1024), 256, 0, stream>>>(ctx);
    context_gemm<<<dim3(B_ * NH, 4), 256, 0, stream>>>(qkv, memkv, rmax, rsum, ctx);
    attn_out    <<<dim3(S_ / 128, B_ * NH), 256, 0, stream>>>(ctx, qsm, outb);
    out_proj_bn <<<dim3(S_ / 128, C_ / 64, B_), 256, 0, stream>>>(outb, wout, bout, gamma, beta, mean, var, y);
}
